// Net_int_2Edges_attention_20272245637596
// MI455X (gfx1250) — matrix operand order repaired and checked by reading, not compiled, not run on hardware
//
#include <hip/hip_runtime.h>
#include <math.h>

// ---------------- problem constants ----------------
#define NN   100000      // nodes (== 6250 * 16, no M-tail)
#define EE   1600000     // edges layer 1/2
#define EE3  800000      // edges layer 3/4 (symmetrized -> 1.6M)
#define D1   64
#define D2   128
#define RRELU_SLOPE 0.2291666666666667f   // (1/8 + 1/3)/2
#define BN_EPS 1e-5f

typedef __attribute__((ext_vector_type(16))) _Float16 v16h;
typedef __attribute__((ext_vector_type(8)))  float    v8f;

__device__ __forceinline__ float rrelu_f(float v){ return v >= 0.f ? v : RRELU_SLOPE * v; }
__device__ __forceinline__ float lrelu02(float v){ return v >= 0.f ? v : 0.2f * v; }
__device__ __forceinline__ float sigm(float v){ return 1.f / (1.f + expf(-v)); }

// ordered-int float atomic max (works with init = -INF)
__device__ __forceinline__ void atomicMaxF(float* addr, float val){
  if (val >= 0.f) atomicMax((int*)addr, __float_as_int(val));
  else            atomicMin((unsigned int*)addr, __float_as_uint(val));
}

// resolve (src,dst) with optional symmetrization: idx >= nBase -> swapped copy
__device__ __forceinline__ void edge_sd(const int* __restrict__ src, const int* __restrict__ dst,
                                        int nBase, int idx, int& s, int& d){
  if (idx < nBase){ s = src[idx]; d = dst[idx]; }
  else            { s = dst[idx - nBase]; d = src[idx - nBase]; }
}

// pack 8 f32 (two float4) into half of a v16h fragment
__device__ __forceinline__ void pack8(v16h& v, int base, float4 lo, float4 hi){
  v[base+0]=(_Float16)lo.x; v[base+1]=(_Float16)lo.y;
  v[base+2]=(_Float16)lo.z; v[base+3]=(_Float16)lo.w;
  v[base+4]=(_Float16)hi.x; v[base+5]=(_Float16)hi.y;
  v[base+6]=(_Float16)hi.z; v[base+7]=(_Float16)hi.w;
}

// ---------------------------------------------------------------------------
// WMMA GEMM:  C[nrows,M] = A[nrows,K] @ W[M,K]^T (+bias) (+RReLU)
// One wave -> 16x64 output tile (4 accumulators share one A fragment).
// Fragments built directly from global (L2-resident) via b128 loads + v_cvt:
//   A 16x32 f16 (ISA 7.12.2): lane half h, row r=lane&15,
//       elements e<8  -> k = e + 8h        (contiguous 8 floats)
//       elements e>=8 -> k = e + 8 + 8h    (contiguous 8 floats)
//   B 32x16 f16: col r, same k order as A (two contiguous 8-float runs of W row)
//   D 16x16 f32: reg t -> M = t + 8h, N = r
// K templated (64/128) so the K-loop fully unrolls -> back-to-back v_wmma.
// ---------------------------------------------------------------------------
template<int K>
__global__ __launch_bounds__(32)
void k_wmma_gemm(const float* __restrict__ A, const float* __restrict__ W,
                 const float* __restrict__ bias, float* __restrict__ C,
                 int M, int act)
{
  const int lane = threadIdx.x;
  const int half = lane >> 4;       // 0 / 1
  const int r    = lane & 15;
  const int row0 = blockIdx.x * 16;
  const int col0 = blockIdx.y * 64;

  v8f acc[4] = {{}, {}, {}, {}};
  const float* arow = A + (size_t)(row0 + r) * K;

  #pragma unroll
  for (int k0 = 0; k0 < K; k0 += 32){
    if (k0 + 32 < K)   // pull next A chunk toward L0 (global_prefetch_b8)
      __builtin_prefetch(arow + k0 + 32, 0, 1);

    // A fragment: two contiguous 8-float runs of this lane's row
    const float* ap = arow + k0 + 8 * half;
    float4 a0 = *(const float4*)(ap);
    float4 a1 = *(const float4*)(ap + 4);
    float4 a2 = *(const float4*)(ap + 16);
    float4 a3 = *(const float4*)(ap + 20);
    v16h av;
    pack8(av, 0, a0, a1);
    pack8(av, 8, a2, a3);

    #pragma unroll
    for (int t = 0; t < 4; ++t){
      // B fragment: two contiguous 8-float runs of weight row (col0 + 16t + r), same k order as A
      const float* wp = W + (size_t)(col0 + t * 16 + r) * K + k0 + 8 * half;
      float4 b0 = *(const float4*)(wp);
      float4 b1 = *(const float4*)(wp + 4);
      float4 b2 = *(const float4*)(wp + 16);
      float4 b3 = *(const float4*)(wp + 20);
      v16h bv;
      pack8(bv, 0, b0, b1);
      pack8(bv, 8, b2, b3);
      acc[t] = __builtin_amdgcn_wmma_f32_16x16x32_f16(false, av, false, bv,
                                                      (short)0, acc[t], false, false);
    }
  }

  #pragma unroll
  for (int t = 0; t < 4; ++t){
    const int col = col0 + t * 16 + r;
    const float bv = bias ? bias[col] : 0.f;
    #pragma unroll
    for (int reg = 0; reg < 8; ++reg){
      int row = row0 + reg + 8 * half;
      float v = acc[t][reg] + bv;
      if (act == 1) v = rrelu_f(v);
      C[(size_t)row * M + col] = v;
    }
  }
}

// ---------------- elementwise / edge kernels ----------------

__global__ void k_fill(float* __restrict__ p, float v, int n){
  int i = blockIdx.x * blockDim.x + threadIdx.x;
  if (i < n) p[i] = v;
}

// out[n,j] = rrelu( sum_k BN(x[n,k]) * W[j,k] + b[j] ),  K = 8, Mout = 64
__global__ void k_node_encode(const float* __restrict__ x,
                              const float* __restrict__ g, const float* __restrict__ bt,
                              const float* __restrict__ m, const float* __restrict__ v,
                              const float* __restrict__ W, const float* __restrict__ b,
                              float* __restrict__ out)
{
  int i = blockIdx.x * blockDim.x + threadIdx.x;
  if (i >= NN * D1) return;
  int n = i / D1, j = i % D1;
  float s = 0.f;
  #pragma unroll
  for (int k = 0; k < 8; ++k){
    float xb = (x[n * 8 + k] - m[k]) * rsqrtf(v[k] + BN_EPS) * g[k] + bt[k];
    s += xb * W[j * 8 + k];
  }
  out[(size_t)n * D1 + j] = rrelu_f(s + b[j]);
}

// a_s[n] = h[n,:]·as ; a_d[n] = h[n,:]·ad
__global__ void k_scores(const float* __restrict__ h, const float* __restrict__ avs,
                         const float* __restrict__ avd, float* __restrict__ a_s,
                         float* __restrict__ a_d, int D)
{
  int n = blockIdx.x * blockDim.x + threadIdx.x;
  if (n >= NN) return;
  float s0 = 0.f, s1 = 0.f;
  const float* hp = h + (size_t)n * D;
  for (int k = 0; k < D; k += 4){
    float4 hv = *(const float4*)(hp + k);
    s0 += hv.x * avs[k] + hv.y * avs[k+1] + hv.z * avs[k+2] + hv.w * avs[k+3];
    s1 += hv.x * avd[k] + hv.y * avd[k+1] + hv.z * avd[k+2] + hv.w * avd[k+3];
  }
  a_s[n] = s0; a_d[n] = s1;
}

__global__ void k_edge_max(const int* __restrict__ src, const int* __restrict__ dst,
                           int nBase, int nTotal, const float* __restrict__ a_s,
                           const float* __restrict__ a_d, float* __restrict__ segm)
{
  int i = blockIdx.x * blockDim.x + threadIdx.x;
  if (i >= nTotal) return;
  int s, d; edge_sd(src, dst, nBase, i, s, d);
  atomicMaxF(&segm[d], lrelu02(a_s[s] + a_d[d]));
}

__global__ void k_edge_expsum(const int* __restrict__ src, const int* __restrict__ dst,
                              int nBase, int nTotal, const float* __restrict__ a_s,
                              const float* __restrict__ a_d, const float* __restrict__ segm,
                              float* __restrict__ ew, float* __restrict__ segs)
{
  int i = blockIdx.x * blockDim.x + threadIdx.x;
  if (i >= nTotal) return;
  int s, d; edge_sd(src, dst, nBase, i, s, d);
  float e = expf(lrelu02(a_s[s] + a_d[d]) - segm[d]);
  ew[i] = e;
  atomicAdd(&segs[d], e);
}

// agg[d,:] += (ew/segs[d]) * h[s,:]   (4 dims per thread, float4 gather)
__global__ void k_edge_agg(const int* __restrict__ src, const int* __restrict__ dst,
                           int nBase, int nTotal, const float* __restrict__ ew,
                           const float* __restrict__ segs, const float* __restrict__ h,
                           float* __restrict__ agg, int D)
{
  int i = blockIdx.x * blockDim.x + threadIdx.x;
  int q = D >> 2;
  if (i >= nTotal * q) return;
  int e = i / q, c = i % q;
  int s, d; edge_sd(src, dst, nBase, e, s, d);
  float alpha = ew[e] / segs[d];
  float4 hv = ((const float4*)(h + (size_t)s * D))[c];
  float* ap = agg + (size_t)d * D + c * 4;
  atomicAdd(ap + 0, alpha * hv.x);
  atomicAdd(ap + 1, alpha * hv.y);
  atomicAdd(ap + 2, alpha * hv.z);
  atomicAdd(ap + 3, alpha * hv.w);
}

__global__ void k_bias_rrelu(float* __restrict__ p, const float* __restrict__ b, int D, int n){
  int i = blockIdx.x * blockDim.x + threadIdx.x;
  if (i < n) p[i] = rrelu_f(p[i] + b[i % D]);
}

// GRU gate: h' = (1-z)*n + z*h  (in place on state)
__global__ void k_gru(const float* __restrict__ gx, const float* __restrict__ gh,
                      float* __restrict__ h, int D)
{
  int i = blockIdx.x * blockDim.x + threadIdx.x;
  if (i >= NN * D) return;
  int n = i / D, d = i % D;
  const float* px = gx + (size_t)n * 3 * D;
  const float* ph = gh + (size_t)n * 3 * D;
  float r  = sigm(px[d] + ph[d]);
  float z  = sigm(px[D + d] + ph[D + d]);
  float nn = tanhf(px[2 * D + d] + r * ph[2 * D + d]);
  float ho = h[i];
  h[i] = (1.f - z) * nn + z * ho;
}

__global__ void k_bn(const float* __restrict__ x, const float* __restrict__ g,
                     const float* __restrict__ bt, const float* __restrict__ m,
                     const float* __restrict__ v, float* __restrict__ o, int D, int n)
{
  int i = blockIdx.x * blockDim.x + threadIdx.x;
  if (i >= n) return;
  int d = i % D;
  o[i] = (x[i] - m[d]) * rsqrtf(v[d] + BN_EPS) * g[d] + bt[d];
}

// Edge readout: one wave per e3 edge; 384 yhat features, wave shfl reduction.
__global__ __launch_bounds__(256)
void k_readout(const int* __restrict__ e3, const float* __restrict__ attr,
               const float* __restrict__ state,
               const float* __restrict__ bg, const float* __restrict__ bb,
               const float* __restrict__ bm, const float* __restrict__ bv,
               const float* __restrict__ lw, const float* __restrict__ lb,
               float* __restrict__ out)
{
  int wave = threadIdx.x >> 5, lane = threadIdx.x & 31;
  int e = blockIdx.x * 8 + wave;
  if (e >= EE3) return;
  int i0 = e3[e], i1 = e3[EE3 + e];
  float a[8];
  #pragma unroll
  for (int k = 0; k < 8; ++k) a[k] = attr[(size_t)e * 8 + k];
  const float* t0p = state + (size_t)i0 * D2;
  const float* t1p = state + (size_t)i1 * D2;
  float total = 0.f;
  #pragma unroll
  for (int it = 0; it < 12; ++it){
    int j  = lane + 32 * it;
    int jj = j & 127;
    float t0 = t0p[jj], t1 = t1p[jj];
    float y = (j < 128) ? 0.5f * (t0 + t1) : ((j < 256) ? t0 * t1 : (t0 - t1) * (t0 - t1));
    y = (y - bm[j]) * rsqrtf(bv[j] + BN_EPS) * bg[j] + bb[j];
    float w = 0.f;
    #pragma unroll
    for (int k = 0; k < 8; ++k) w += a[k] * lw[j * 8 + k];
    total += y * w;
  }
  #pragma unroll
  for (int off = 16; off; off >>= 1) total += __shfl_xor(total, off, 32);
  if (lane == 0){
    float bs = 0.f;
    #pragma unroll
    for (int k = 0; k < 8; ++k) bs += a[k] * lb[k];
    out[e] = total + bs;
  }
}

// ---------------- host-side orchestration ----------------

static inline int cdiv(int a, int b){ return (a + b - 1) / b; }

extern "C" void kernel_launch(void* const* d_in, const int* in_sizes, int n_in,
                              void* d_out, int out_size, void* d_ws, size_t ws_size,
                              hipStream_t stream)
{
  (void)in_sizes; (void)n_in; (void)out_size; (void)ws_size;
  auto F = [&](int i){ return (const float*)d_in[i]; };

  const float* x     = F(0);
  const int*   ei    = (const int*)d_in[1];   // src = ei, dst = ei + EE
  const int*   ei3   = (const int*)d_in[2];   // src = ei3, dst = ei3 + EE3
  const float* attr3 = F(3);
  // params, setup_inputs() insertion order, tuples flattened:
  const float *bnx_g=F(4), *bnx_b=F(5), *bnx_m=F(6), *bnx_v=F(7);
  const float *linW=F(8), *linB=F(9);
  const float *W1=F(10), *as1=F(11), *ad1=F(12), *b1=F(13);
  const float *g1Wih=F(14), *g1Whh=F(15), *g1bih=F(16), *g1bhh=F(17);
  const float *bnc_g=F(18), *bnc_b=F(19), *bnc_m=F(20), *bnc_v=F(21);
  const float *c1W=F(22), *c1b=F(23), *c2W=F(24), *c2b=F(25);
  const float *W2=F(26), *as2=F(27), *ad2=F(28), *b2=F(29);
  const float *g2Wih=F(30), *g2Whh=F(31), *g2bih=F(32), *g2bhh=F(33);
  const float *bno_g=F(34), *bno_b=F(35), *bno_m=F(36), *bno_v=F(37);
  const float *lw=F(38), *lb=F(39);
  float* out = (float*)d_out;

  // workspace carve-out (256B aligned regions)
  char* w = (char*)d_ws;
  auto carve = [&](size_t bytes) -> float* {
    float* p = (float*)w; w += (bytes + 255) & ~(size_t)255; return p;
  };
  float* state = carve((size_t)NN * D2 * 4);       // node state (h == out)
  float* bufA  = carve((size_t)NN * D2 * 4);       // GAT h-projection / BN tmp
  float* bufB  = carve((size_t)NN * D2 * 4);       // aggregation / GRU input m
  float* gx    = carve((size_t)NN * 3 * D2 * 4);
  float* gh    = carve((size_t)NN * 3 * D2 * 4);
  float* a_s   = carve((size_t)NN * 4);
  float* a_d   = carve((size_t)NN * 4);
  float* segm  = carve((size_t)NN * 4);
  float* segs  = carve((size_t)NN * 4);
  float* ew    = carve((size_t)EE * 4);            // per-edge exp weights

  const int T = 256;
  // M must be a multiple of 64 here (64/128/192/384 all are)
  auto gemm = [&](const float* A, const float* W_, const float* bias, float* C,
                  int K, int M, int act){
    dim3 g(NN / 16, M / 64);
    if (K == 64) k_wmma_gemm<64 ><<<g, 32, 0, stream>>>(A, W_, bias, C, M, act);
    else         k_wmma_gemm<128><<<g, 32, 0, stream>>>(A, W_, bias, C, M, act);
  };

  // GAT + GRU layer pair
  auto gat_gru = [&](const int* src, const int* dst, int nBase, int nTotal, int D,
                     const float* Wg, const float* avs, const float* avd, const float* bg,
                     const float* Wih, const float* Whh, const float* bih, const float* bhh){
    gemm(state, Wg, nullptr, bufA, D, D, 0);                               // h = out @ W^T
    k_scores<<<cdiv(NN, T), T, 0, stream>>>(bufA, avs, avd, a_s, a_d, D);
    k_fill<<<cdiv(NN, T), T, 0, stream>>>(segm, -INFINITY, NN);
    k_fill<<<cdiv(NN, T), T, 0, stream>>>(segs, 0.f, NN);
    k_fill<<<cdiv(NN * D, T), T, 0, stream>>>(bufB, 0.f, NN * D);
    k_edge_max<<<cdiv(nTotal, T), T, 0, stream>>>(src, dst, nBase, nTotal, a_s, a_d, segm);
    k_edge_expsum<<<cdiv(nTotal, T), T, 0, stream>>>(src, dst, nBase, nTotal, a_s, a_d, segm, ew, segs);
    k_edge_agg<<<cdiv(nTotal * (D / 4), T), T, 0, stream>>>(src, dst, nBase, nTotal, ew, segs, bufA, bufB, D);
    k_bias_rrelu<<<cdiv(NN * D, T), T, 0, stream>>>(bufB, bg, D, NN * D);  // m = rrelu(agg + b)
    gemm(bufB,  Wih, bih, gx, D, 3 * D, 0);
    gemm(state, Whh, bhh, gh, D, 3 * D, 0);
    k_gru<<<cdiv(NN * D, T), T, 0, stream>>>(gx, gh, state, D);            // state <- GRU
  };

  // 1) node encoder: BN -> Linear(8->64) -> RReLU
  k_node_encode<<<cdiv(NN * D1, T), T, 0, stream>>>(x, bnx_g, bnx_b, bnx_m, bnx_v,
                                                    linW, linB, state);
  // 2) two GAT(64)+GRU(64) rounds on edge_index
  for (int it = 0; it < 2; ++it)
    gat_gru(ei, ei + EE, EE, EE, D1, W1, as1, ad1, b1, g1Wih, g1Whh, g1bih, g1bhh);

  // 3) lin_covert: BN -> Linear(64->128) -> RReLU -> Linear(128->128) -> RReLU
  k_bn<<<cdiv(NN * D1, T), T, 0, stream>>>(state, bnc_g, bnc_b, bnc_m, bnc_v, bufA, D1, NN * D1);
  gemm(bufA, c1W, c1b, bufB, D1, D2, 1);
  gemm(bufB, c2W, c2b, state, D2, D2, 1);

  // 4) two GAT(128)+GRU(128) rounds on symmetrized edge_index3
  for (int it = 0; it < 2; ++it)
    gat_gru(ei3, ei3 + EE3, EE3, 2 * EE3, D2, W2, as2, ad2, b2, g2Wih, g2Whh, g2bih, g2bhh);

  // 5) edge readout
  k_readout<<<cdiv(EE3, 8), 256, 0, stream>>>(ei3, attr3, state,
                                              bno_g, bno_b, bno_m, bno_v, lw, lb, out);
}
